// BitLinearSTE_57492432224590
// MI455X (gfx1250) — matrix operand order repaired and checked by reading, not compiled, not run on hardware
//
#include <hip/hip_runtime.h>
#include <hip/hip_bf16.h>
#include <stdint.h>

typedef __attribute__((ext_vector_type(16))) __bf16 v16bf;
typedef __attribute__((ext_vector_type(8)))  __bf16 v8bf;
typedef __attribute__((ext_vector_type(8)))  float  v8f;

#define TILE_M 128
#define TILE_N 128
#define TILE_K 32
#define LDSS   40   // halves per row = 80 B: rows 16B-aligned, 20-dword stride => conflict-free ds_load_b128
#define USE_ASYNC_LDS 1

// ---------------------------------------------------------------------------
// Wait helper for async global->LDS copies (ASYNCcnt)
// ---------------------------------------------------------------------------
__device__ __forceinline__ void wait_async_copies() {
#if USE_ASYNC_LDS
#if __has_builtin(__builtin_amdgcn_s_wait_asynccnt)
    __builtin_amdgcn_s_wait_asynccnt(0);
#else
    asm volatile("s_wait_asynccnt 0x0" ::: "memory");
#endif
#endif
}

// ===========================================================================
// Preprocess kernels (run once per launch; overwrite their d_ws regions fully)
//   x = hi + lo (bf16 split-precision, ~16 mantissa bits total)
//   sign(W) in {-1,0,+1} is exact in bf16
// ===========================================================================
__global__ __launch_bounds__(256)
void prep_x_split(const float* __restrict__ X, __bf16* __restrict__ Xhi,
                  __bf16* __restrict__ Xlo, size_t n)
{
    size_t i = ((size_t)blockIdx.x * 256 + threadIdx.x) * 8;
    if (i >= n) return;
    const float4 a = *(const float4*)(X + i);
    const float4 b = *(const float4*)(X + i + 4);
    const float v[8] = {a.x, a.y, a.z, a.w, b.x, b.y, b.z, b.w};
    v8bf hi, lo;
#pragma unroll
    for (int j = 0; j < 8; ++j) {
        __bf16 h = (__bf16)v[j];
        hi[j] = h;
        lo[j] = (__bf16)(v[j] - (float)h);
    }
    *(v8bf*)(Xhi + i) = hi;
    *(v8bf*)(Xlo + i) = lo;
}

__global__ __launch_bounds__(256)
void prep_w_sign(const float* __restrict__ W, __bf16* __restrict__ Ws, size_t n)
{
    size_t i = ((size_t)blockIdx.x * 256 + threadIdx.x) * 8;
    if (i >= n) return;
    const float4 a = *(const float4*)(W + i);
    const float4 b = *(const float4*)(W + i + 4);
    const float v[8] = {a.x, a.y, a.z, a.w, b.x, b.y, b.z, b.w};
    v8bf s;
#pragma unroll
    for (int j = 0; j < 8; ++j)
        s[j] = (v[j] > 0.f) ? (__bf16)1.0f : ((v[j] < 0.f) ? (__bf16)-1.0f : (__bf16)0.0f);
    *(v8bf*)(Ws + i) = s;
}

// ===========================================================================
// Main GEMM: pure bf16 tiles, async global->LDS double-buffered pipeline.
// y[m,n] = sum_k (xhi+xlo)[m,k] * sgn[n,k]
// ===========================================================================
__global__ __launch_bounds__(256)
void bitlinear_gemm_pre(const __bf16* __restrict__ Xhi,
                        const __bf16* __restrict__ Xlo,
                        const __bf16* __restrict__ Wsg,
                        float* __restrict__ Y, int M, int N, int K)
{
    __shared__ __bf16 sAhi[2][TILE_M * LDSS];
    __shared__ __bf16 sAlo[2][TILE_M * LDSS];
    __shared__ __bf16 sW  [2][TILE_N * LDSS];

    const int tid   = threadIdx.x;
    const int lane  = tid & 31;
    const int wave  = tid >> 5;
    const int waveM = wave >> 2;   // 0..1
    const int waveN = wave & 3;    // 0..3
    const int bM    = blockIdx.y * TILE_M;
    const int bN    = blockIdx.x * TILE_N;

    v8f acc[4][2];
#pragma unroll
    for (int mt = 0; mt < 4; ++mt)
#pragma unroll
        for (int nt = 0; nt < 2; ++nt)
            acc[mt][nt] = (v8f){0.f,0.f,0.f,0.f,0.f,0.f,0.f,0.f};

    // 3 tiles of 128x32 bf16 (8 KB each): 512 x 16B chunks per tile, 2 per thread.
    auto issueCopies = [&](int p, int k0) {
#pragma unroll
        for (int j = 0; j < 2; ++j) {
            int chunk = tid + j * 256;       // 0..511
            int row   = chunk >> 2;          // 0..127
            int c     = chunk & 3;           // 16B chunk within the 64B row
            const __bf16* ga = Xhi + (size_t)(bM + row) * K + k0 + c * 8;
            const __bf16* gl = Xlo + (size_t)(bM + row) * K + k0 + c * 8;
            const __bf16* gw = Wsg + (size_t)(bN + row) * K + k0 + c * 8;
#if USE_ASYNC_LDS
            uint32_t oh = (uint32_t)(uintptr_t)&sAhi[p][row * LDSS + c * 8];
            uint32_t ol = (uint32_t)(uintptr_t)&sAlo[p][row * LDSS + c * 8];
            uint32_t ow = (uint32_t)(uintptr_t)&sW  [p][row * LDSS + c * 8];
            asm volatile("global_load_async_to_lds_b128 %0, %1, off" :: "v"(oh), "v"(ga) : "memory");
            asm volatile("global_load_async_to_lds_b128 %0, %1, off" :: "v"(ol), "v"(gl) : "memory");
            asm volatile("global_load_async_to_lds_b128 %0, %1, off" :: "v"(ow), "v"(gw) : "memory");
#else
            *(v8bf*)&sAhi[p][row * LDSS + c * 8] = *(const v8bf*)ga;
            *(v8bf*)&sAlo[p][row * LDSS + c * 8] = *(const v8bf*)gl;
            *(v8bf*)&sW  [p][row * LDSS + c * 8] = *(const v8bf*)gw;
#endif
        }
    };

    struct V2 { v8bf a, b; };
    const int rsel = lane >> 4;
    const int lm   = lane & 15;
    const int kh   = rsel * 8;

    const int nk = K / TILE_K;
    issueCopies(0, 0);
    wait_async_copies();
    __syncthreads();

    int p = 0;
    for (int i = 0; i < nk; ++i) {
        if (i + 1 < nk) issueCopies(p ^ 1, (i + 1) * TILE_K);   // DMA overlaps WMMA below

        // B fragments: lane = column, K {kh..kh+7} | {16+kh..16+kh+7}
        v16bf bfrag[2];
#pragma unroll
        for (int nt = 0; nt < 2; ++nt) {
            const __bf16* bb = &sW[p][(waveN * 32 + nt * 16 + lm) * LDSS + rsel * 8];
            V2 tb = { *(const v8bf*)bb, *(const v8bf*)(bb + 16) };
            bfrag[nt] = __builtin_bit_cast(v16bf, tb);
        }
#pragma unroll
        for (int mt = 0; mt < 4; ++mt) {
            int r = waveM * 64 + mt * 16 + lm;
            const __bf16* bh = &sAhi[p][r * LDSS + kh];
            const __bf16* bl = &sAlo[p][r * LDSS + kh];
            V2 th = { *(const v8bf*)bh, *(const v8bf*)(bh + 16) };  // K {kh..kh+7} | {16+kh..}
            V2 tl = { *(const v8bf*)bl, *(const v8bf*)(bl + 16) };
            v16bf ah = __builtin_bit_cast(v16bf, th);
            v16bf al = __builtin_bit_cast(v16bf, tl);
#pragma unroll
            for (int nt = 0; nt < 2; ++nt) {
                acc[mt][nt] = __builtin_amdgcn_wmma_f32_16x16x32_bf16(
                    false, ah, false, bfrag[nt], (short)0, acc[mt][nt], false, false);
                acc[mt][nt] = __builtin_amdgcn_wmma_f32_16x16x32_bf16(
                    false, al, false, bfrag[nt], (short)0, acc[mt][nt], false, false);
            }
        }
        wait_async_copies();
        __syncthreads();
        p ^= 1;
    }

    // C/D layout: VGPR r -> M = r + (lane[4] ? 8 : 0), N = lane&15
#pragma unroll
    for (int mt = 0; mt < 4; ++mt) {
        int rowBase = bM + waveM * 64 + mt * 16 + rsel * 8;
#pragma unroll
        for (int nt = 0; nt < 2; ++nt) {
            int col = bN + waveN * 32 + nt * 16 + lm;
#pragma unroll
            for (int r = 0; r < 8; ++r)
                Y[(size_t)(rowBase + r) * N + col] = acc[mt][nt][r];
        }
    }
}

// ===========================================================================
// Fallback: fused convert-in-loop kernel (used only if d_ws is too small)
// ===========================================================================
#define FLDSS 48
__global__ __launch_bounds__(256)
void bitlinear_ste_fused(const float* __restrict__ X, const float* __restrict__ W,
                         float* __restrict__ Y, int M, int N, int K)
{
    __shared__ __bf16 sAhi[TILE_M * FLDSS];
    __shared__ __bf16 sAlo[TILE_M * FLDSS];
    __shared__ __bf16 sW  [TILE_N * FLDSS];

    const int tid = threadIdx.x, lane = tid & 31, wave = tid >> 5;
    const int waveM = wave >> 2, waveN = wave & 3;
    const int bM = blockIdx.y * TILE_M, bN = blockIdx.x * TILE_N;

    v8f acc[4][2];
#pragma unroll
    for (int mt = 0; mt < 4; ++mt)
#pragma unroll
        for (int nt = 0; nt < 2; ++nt)
            acc[mt][nt] = (v8f){0.f,0.f,0.f,0.f,0.f,0.f,0.f,0.f};

    float4 aReg[4], wReg[4];
    auto loadTiles = [&](int k0) {
#pragma unroll
        for (int i = 0; i < 4; ++i) {
            int f4 = tid + i * 256, row = f4 >> 3, kk = (f4 & 7) << 2;
            aReg[i] = *(const float4*)(X + (size_t)(bM + row) * K + k0 + kk);
            wReg[i] = *(const float4*)(W + (size_t)(bN + row) * K + k0 + kk);
        }
    };
    auto stageTiles = [&]() {
#pragma unroll
        for (int i = 0; i < 4; ++i) {
            int f4 = tid + i * 256, row = f4 >> 3, kk = (f4 & 7) << 2;
            const float av[4] = {aReg[i].x, aReg[i].y, aReg[i].z, aReg[i].w};
            const float wv[4] = {wReg[i].x, wReg[i].y, wReg[i].z, wReg[i].w};
#pragma unroll
            for (int j = 0; j < 4; ++j) {
                __bf16 h = (__bf16)av[j];
                sAhi[row * FLDSS + kk + j] = h;
                sAlo[row * FLDSS + kk + j] = (__bf16)(av[j] - (float)h);
                sW[row * FLDSS + kk + j] = (wv[j] > 0.f) ? (__bf16)1.0f
                                          : ((wv[j] < 0.f) ? (__bf16)-1.0f : (__bf16)0.0f);
            }
        }
    };

    struct V2 { v8bf a, b; };
    loadTiles(0);
    for (int k0 = 0; k0 < K; k0 += TILE_K) {
        stageTiles();
        __syncthreads();
        if (k0 + TILE_K < K) loadTiles(k0 + TILE_K);

        const int rsel = lane >> 4, lm = lane & 15, kh = rsel * 8;
        v16bf bfrag[2];
#pragma unroll
        for (int nt = 0; nt < 2; ++nt) {
            const __bf16* bb = &sW[(waveN * 32 + nt * 16 + lm) * FLDSS + rsel * 8];
            V2 tb = { *(const v8bf*)bb, *(const v8bf*)(bb + 16) };
            bfrag[nt] = __builtin_bit_cast(v16bf, tb);
        }
#pragma unroll
        for (int mt = 0; mt < 4; ++mt) {
            int r = waveM * 64 + mt * 16 + lm;
            const __bf16* bh = &sAhi[r * FLDSS + kh];
            const __bf16* bl = &sAlo[r * FLDSS + kh];
            V2 th = { *(const v8bf*)bh, *(const v8bf*)(bh + 16) };
            V2 tl = { *(const v8bf*)bl, *(const v8bf*)(bl + 16) };
            v16bf ah = __builtin_bit_cast(v16bf, th);
            v16bf al = __builtin_bit_cast(v16bf, tl);
#pragma unroll
            for (int nt = 0; nt < 2; ++nt) {
                acc[mt][nt] = __builtin_amdgcn_wmma_f32_16x16x32_bf16(
                    false, ah, false, bfrag[nt], (short)0, acc[mt][nt], false, false);
                acc[mt][nt] = __builtin_amdgcn_wmma_f32_16x16x32_bf16(
                    false, al, false, bfrag[nt], (short)0, acc[mt][nt], false, false);
            }
        }
        __syncthreads();
    }

    const int lm = lane & 15, rsel = lane >> 4;
#pragma unroll
    for (int mt = 0; mt < 4; ++mt) {
        int rowBase = bM + waveM * 64 + mt * 16 + rsel * 8;
#pragma unroll
        for (int nt = 0; nt < 2; ++nt) {
            int col = bN + waveN * 32 + nt * 16 + lm;
#pragma unroll
            for (int r = 0; r < 8; ++r)
                Y[(size_t)(rowBase + r) * N + col] = acc[mt][nt][r];
        }
    }
}

extern "C" void kernel_launch(void* const* d_in, const int* in_sizes, int n_in,
                              void* d_out, int out_size, void* d_ws, size_t ws_size,
                              hipStream_t stream) {
    (void)n_in; (void)out_size;
    const float* X = (const float*)d_in[0];   // (4,2048,4096) f32
    const float* W = (const float*)d_in[1];   // (4096,4096)  f32
    float*       Y = (float*)d_out;

    const int K = 4096, N = 4096;
    const int M = in_sizes[0] / K;            // 8192
    const size_t MK = (size_t)M * K, NK = (size_t)N * K;
    const size_t need = MK * 4 + NK * 2;      // Xhi+Xlo (bf16) + sign(W) (bf16)

    dim3 grid(N / TILE_N, M / TILE_M);
    if (ws_size >= need) {
        __bf16* Xhi = (__bf16*)d_ws;
        __bf16* Xlo = Xhi + MK;
        __bf16* Wsg = Xlo + MK;
        prep_x_split<<<(unsigned)(MK / 2048), 256, 0, stream>>>(X, Xhi, Xlo, MK);
        prep_w_sign <<<(unsigned)(NK / 2048), 256, 0, stream>>>(W, Wsg, NK);
        bitlinear_gemm_pre<<<grid, 256, 0, stream>>>(Xhi, Xlo, Wsg, Y, M, N, K);
    } else {
        bitlinear_ste_fused<<<grid, 256, 0, stream>>>(X, W, Y, M, N, K);
    }
}
